// GCN_84456236909326
// MI455X (gfx1250) — matrix operand order repaired and checked by reading, not compiled, not run on hardware
//
#include <hip/hip_runtime.h>
#include <hip/hip_bf16.h>
#include <stdint.h>

typedef __bf16 bf16_t;
typedef __attribute__((ext_vector_type(16))) __bf16 v16bf;
typedef __attribute__((ext_vector_type(8)))  float  v8f;
typedef __attribute__((ext_vector_type(4)))  float  v4f;

#define N_DIM 16384
#define D     128           // D_IN == D_HID == D_OUT
#define NKC   (N_DIM / 32)  // 512 k-chunks of 32
#define NT    (D / 16)      // 8 column tiles
#define WPB   4             // waves per block
#define BM    (WPB * 16)    // rows per block
#define CHUNK_BYTES 8192    // one packed B chunk: 32x128 bf16

// ---------------- pack X: f32 [N,128] -> bf16 B-fragment layout ----------------
// frag f = kc*8+nt; lane holds 16 bf16: element i = X[kc*32 + (lane/16)*16 + i][nt*16 + lane%16]
__global__ void gcn_pack_x(const float* __restrict__ X, bf16_t* __restrict__ Xp) {
    int t    = blockIdx.x * blockDim.x + threadIdx.x;
    int lane = t & 31;
    int frag = t >> 5;
    if (frag >= NKC * NT) return;
    int kc = frag >> 3, nt = frag & 7;
    int col  = nt * 16 + (lane & 15);
    int krow = kc * 32 + (lane >> 4) * 16;
    bf16_t* dst = Xp + (size_t)frag * 512 + (size_t)lane * 16;
#pragma unroll
    for (int i = 0; i < 16; ++i)
        dst[i] = (bf16_t)X[(size_t)(krow + i) * D + col];
}

// ---------------- pack W1/W2 (transposed): Wt[k][n] = W[n][k] -------------------
__global__ void gcn_pack_w(const float* __restrict__ W1, const float* __restrict__ W2,
                           bf16_t* __restrict__ W1p, bf16_t* __restrict__ W2p) {
    int t    = blockIdx.x * blockDim.x + threadIdx.x;
    int lane = t & 31;
    int frag = t >> 5;                 // 0..63 (32 per matrix)
    if (frag >= 64) return;
    const float* W  = (frag < 32) ? W1 : W2;
    bf16_t*      Wp = (frag < 32) ? W1p : W2p;
    int f  = frag & 31;
    int kc = f >> 3, nt = f & 7;
    int n  = nt * 16 + (lane & 15);
    int k0 = kc * 32 + (lane >> 4) * 16;
    bf16_t* dst = Wp + (size_t)f * 512 + (size_t)lane * 16;
#pragma unroll
    for (int i = 0; i < 16; ++i)
        dst[i] = (bf16_t)W[(size_t)n * D + k0 + i];
}

// ---------------- main layer: out = act((A @ B) @ W^T) --------------------------
// B chunks staged in LDS via async global->LDS copies (ASYNCcnt), double-buffered.
// A loads software-pipelined one chunk ahead; B frags pipelined 2-deep from LDS.
// Buffer swap uses a raw split barrier (no implicit fence) so A loads pipeline across it.
template <bool RELU_PACK>
__global__ __launch_bounds__(WPB * 32)
void gcn_layer(const float* __restrict__ A,
               const bf16_t* __restrict__ Bp,   // packed [NKC][NT][32][16]
               const bf16_t* __restrict__ Wp,   // packed [4][NT][32][16]
               bf16_t* __restrict__ Hp,         // packed bf16 out (RELU_PACK)
               float* __restrict__ Out)         // f32 out (!RELU_PACK)
{
    const int lane = threadIdx.x & 31;
    const int wid  = threadIdx.x >> 5;     // 0..WPB-1
    const int hl   = lane >> 4;            // lane half (0/1)
    const int lr   = lane & 15;
    const int r0   = blockIdx.x * BM + wid * 16;   // wave's 16 rows

    __shared__ __align__(16) bf16_t Bsh[2][CHUNK_BYTES / 2];  // double-buffered B chunk
    __shared__ __align__(16) float  Psh[WPB][16][132];

    // each wave async-copies its quarter of the chunk: 4 x (32 lanes x 16B) = 2 KB
    const unsigned sub  = (unsigned)(wid * 2048 + lane * 16);
    const unsigned lds0 = (unsigned)(uintptr_t)&Bsh[0][0];
    const unsigned lds1 = (unsigned)(uintptr_t)&Bsh[1][0];

    // A-frag k order used here (same as B packing): lanes 0-15 hold K {0..15}, lanes 16-31 hold K {16..31}
    const float* arow = A + (size_t)(r0 + lr) * N_DIM + hl * 16;

    // ---- prologue: stage chunk 0 into buffer 0 ----
    {
        unsigned la   = lds0 + sub;
        unsigned goff = sub;
        asm volatile(
            "global_load_async_to_lds_b128 %0, %1, %2 offset:0\n\t"
            "global_load_async_to_lds_b128 %0, %1, %2 offset:512\n\t"
            "global_load_async_to_lds_b128 %0, %1, %2 offset:1024\n\t"
            "global_load_async_to_lds_b128 %0, %1, %2 offset:1536\n\t"
            "s_wait_asynccnt 0"
            :: "v"(la), "v"(goff), "s"(Bp) : "memory");
    }
    __syncthreads();

    v8f acc[NT] = {};

    // A pipeline: preload chunk 0's A elements
    v4f a0 = *(const v4f*)(arow + 0);
    v4f a1 = *(const v4f*)(arow + 4);
    v4f a2 = *(const v4f*)(arow + 8);
    v4f a3 = *(const v4f*)(arow + 12);
    arow += 32;

#pragma unroll 1
    for (int kc = 0; kc < NKC; ++kc) {
        const unsigned cur = (unsigned)kc & 1u;

        // kick off async stage of chunk kc+1 into the other buffer
        // (last iteration: re-stage the final chunk into the dead buffer — branch-free)
        {
            unsigned nk   = (kc + 1 < NKC) ? (unsigned)(kc + 1) : (unsigned)(NKC - 1);
            unsigned la   = (cur ? lds0 : lds1) + sub;
            unsigned goff = nk * CHUNK_BYTES + sub;
            asm volatile(
                "global_load_async_to_lds_b128 %0, %1, %2 offset:0\n\t"
                "global_load_async_to_lds_b128 %0, %1, %2 offset:512\n\t"
                "global_load_async_to_lds_b128 %0, %1, %2 offset:1024\n\t"
                "global_load_async_to_lds_b128 %0, %1, %2 offset:1536"
                :: "v"(la), "v"(goff), "s"(Bp) : "memory");
        }

        // convert A loaded one chunk ago (its latency is already paid)
        v16bf af;
#pragma unroll
        for (int i = 0; i < 4; ++i) {
            af[i]      = (bf16_t)a0[i];
            af[4 + i]  = (bf16_t)a1[i];
            af[8 + i]  = (bf16_t)a2[i];
            af[12 + i] = (bf16_t)a3[i];
        }

        // issue A loads for chunk kc+1; they hide under this chunk's WMMAs
        // (and survive across the barrier below, which has no hardware fence)
        if (kc + 1 < NKC) {
            a0 = *(const v4f*)(arow + 0);
            a1 = *(const v4f*)(arow + 4);
            a2 = *(const v4f*)(arow + 8);
            a3 = *(const v4f*)(arow + 12);
            __builtin_prefetch(arow + 1024, 0, 0);   // ~32 chunks ahead into GL2
            arow += 32;
        }

        // B frags from LDS, 2-deep pipelined so each WMMA waits on a load issued one WMMA earlier
        const bf16_t* bsh = &Bsh[cur][lane * 16];
        v16bf b0 = *(const v16bf*)(bsh);
#pragma unroll
        for (int nt = 0; nt < NT; nt += 2) {
            v16bf b1 = *(const v16bf*)(bsh + (nt + 1) * 512);
            acc[nt] = __builtin_amdgcn_wmma_f32_16x16x32_bf16(
                false, af, false, b0, (short)0, acc[nt], false, false);
            if (nt + 2 < NT)
                b0 = *(const v16bf*)(bsh + (nt + 2) * 512);
            acc[nt + 1] = __builtin_amdgcn_wmma_f32_16x16x32_bf16(
                false, af, false, b1, (short)0, acc[nt + 1], false, false);
        }

        // Buffer swap: my async copies must have landed; all my ds reads of the old
        // buffer were already consumed by the WMMAs above. Raw split barrier only —
        // no loadcnt/dscnt drain, so the pipelined A loads stay in flight.
        asm volatile(
            "s_wait_asynccnt 0\n\t"
            "s_barrier_signal -1\n\t"
            "s_barrier_wait -1"
            ::: "memory");
    }

    // ---- stage P = (A@B) tile [16 x 128] in LDS (f32), re-fragment as A-operand ----
    float (*P)[132] = Psh[wid];
#pragma unroll
    for (int nt = 0; nt < NT; ++nt)
#pragma unroll
        for (int j = 0; j < 8; ++j)
            P[j + 8 * hl][nt * 16 + lr] = acc[nt][j];   // C/D layout: M = j + 8*half, N = lane%16
    __syncthreads();

    v8f acc2[NT] = {};
    const bf16_t* wptr = Wp + (size_t)lane * 16;
#pragma unroll
    for (int kc = 0; kc < 4; ++kc) {
        const float* prow = &P[lr][kc * 32 + hl * 16];
        v4f p0 = *(const v4f*)(prow + 0);
        v4f p1 = *(const v4f*)(prow + 4);
        v4f p2 = *(const v4f*)(prow + 8);
        v4f p3 = *(const v4f*)(prow + 12);
        v16bf pf;
#pragma unroll
        for (int i = 0; i < 4; ++i) {
            pf[i]      = (bf16_t)p0[i];
            pf[4 + i]  = (bf16_t)p1[i];
            pf[8 + i]  = (bf16_t)p2[i];
            pf[12 + i] = (bf16_t)p3[i];
        }
#pragma unroll
        for (int nt = 0; nt < NT; ++nt) {
            v16bf wf = *(const v16bf*)(wptr + (size_t)(kc * NT + nt) * 512);
            acc2[nt] = __builtin_amdgcn_wmma_f32_16x16x32_bf16(
                false, pf, false, wf, (short)0, acc2[nt], false, false);
        }
    }

    if (RELU_PACK) {
        __syncthreads();
        // ReLU + restage in LDS, then emit bf16 in packed-B layout for next layer
#pragma unroll
        for (int nt = 0; nt < NT; ++nt)
#pragma unroll
            for (int j = 0; j < 8; ++j) {
                float v = acc2[nt][j];
                P[j + 8 * hl][nt * 16 + lr] = v > 0.0f ? v : 0.0f;
            }
        __syncthreads();
        const int kcg = r0 >> 5;           // this wave's rows = half h0 of k-chunk kcg
        const int h0  = (r0 >> 4) & 1;
        if (hl == h0) {
#pragma unroll
            for (int nt = 0; nt < NT; ++nt) {
                bf16_t* dst = Hp + ((size_t)(kcg * NT + nt) * 32 + lane) * 16;
#pragma unroll
                for (int i = 0; i < 16; ++i)
                    dst[i] = (bf16_t)P[i][nt * 16 + lr];
            }
        }
    } else {
#pragma unroll
        for (int nt = 0; nt < NT; ++nt)
#pragma unroll
            for (int j = 0; j < 8; ++j)
                Out[(size_t)(r0 + j + 8 * hl) * D + nt * 16 + lr] = acc2[nt][j];
    }
}

extern "C" void kernel_launch(void* const* d_in, const int* in_sizes, int n_in,
                              void* d_out, int out_size, void* d_ws, size_t ws_size,
                              hipStream_t stream) {
    const float* A  = (const float*)d_in[0];
    const float* X  = (const float*)d_in[1];
    const float* W1 = (const float*)d_in[2];
    const float* W2 = (const float*)d_in[3];
    float* out = (float*)d_out;

    char* ws = (char*)d_ws;
    bf16_t* Xp  = (bf16_t*)(ws);                               // 4 MB
    bf16_t* Hp  = (bf16_t*)(ws + (4u << 20));                  // 4 MB
    bf16_t* W1p = (bf16_t*)(ws + (8u << 20));                  // 32 KB
    bf16_t* W2p = (bf16_t*)(ws + (8u << 20) + (64u << 10));    // 32 KB

    gcn_pack_x<<<(NKC * NT * 32) / 256, 256, 0, stream>>>(X, Xp);
    gcn_pack_w<<<8, 256, 0, stream>>>(W1, W2, W1p, W2p);
    gcn_layer<true ><<<N_DIM / BM, WPB * 32, 0, stream>>>(A, Xp, W1p, Hp, nullptr);
    gcn_layer<false><<<N_DIM / BM, WPB * 32, 0, stream>>>(A, Hp, W2p, nullptr, out);
}
